// Attention_82703890252107
// MI455X (gfx1250) — compile-verified, matrix operand order checked by reading, not run on hardware
//
#include <hip/hip_runtime.h>
#include <hip/hip_bf16.h>
#include <stdint.h>

// ---------------------------------------------------------------------------
// MHA block for MI455X (gfx1250, wave32).  All GEMMs via v_wmma_f32_16x16x32_bf16.
//   x[4,2048,1024] @ w_qkv[1024,3072] -> Q,K,V bf16 [B,H,S,64] (Q pre-scaled)
//   flash attention per (b,h) -> O bf16 [B,H,S,64]   (Q/K tiles fetched with
//       global_load_async_to_lds_b128, ASYNCcnt-tracked)
//   O (viewed [8192,1024]) @ w_out[1024,1024] + b_out -> out f32
// ---------------------------------------------------------------------------

typedef __attribute__((ext_vector_type(16))) __bf16 v16bf;
typedef __attribute__((ext_vector_type(8)))  float  v8f;

#define HEADS 16
#define HEAD_DIM 64
#define SEQ 2048
#define BATCH 4
#define DIM 1024
#define ROWS (BATCH * SEQ)            // 8192

union AF { v16bf v; unsigned u[8]; };
union CF { v8f   v; float    f[8]; };

static __device__ __forceinline__ unsigned short f2bf(float f) {
  unsigned u = __float_as_uint(f);
  unsigned r = u + 0x7FFFu + ((u >> 16) & 1u);   // round-to-nearest-even
  return (unsigned short)(r >> 16);
}

// Async copy of 16B (and 16B at +16) from global to LDS, per lane.
// ISA 15.18.3 op 98; the instruction offset is added to both the LDS and global address.
static __device__ __forceinline__ void async_copy32(unsigned lds_addr,
                                                    const void* gptr) {
  asm volatile("global_load_async_to_lds_b128 %0, %1, off"
               :: "v"(lds_addr), "v"(gptr) : "memory");
  asm volatile("global_load_async_to_lds_b128 %0, %1, off offset:16"
               :: "v"(lds_addr), "v"(gptr) : "memory");
}
static __device__ __forceinline__ void wait_async0() {
  asm volatile("s_wait_asynccnt 0" ::: "memory");
}

// Load a 16x32 bf16 fragment (A or B^T layout) from LDS.
// rowptr points at the lane's matrix row (A: M = lane&15; B: N = lane&15),
// at the start of the 32-element K slice.  ISA 7.12.2 layout:
// lanes 0-15 hold K pairs {0..7,16..23}; lanes 16-31 hold {8..15,24..31}.
static __device__ __forceinline__ v16bf frag32(const unsigned short* rowptr, int lane) {
  AF f;
  const unsigned* p = (const unsigned*)rowptr;
  int h = (lane >> 4) * 4;
#pragma unroll
  for (int j = 0; j < 4; ++j) {
    f.u[j]     = p[h + j];
    f.u[4 + j] = p[8 + h + j];
  }
  return f.v;
}

static __device__ __forceinline__ v8f wmma_bf16(v16bf a, v16bf b, v8f c) {
  return __builtin_amdgcn_wmma_f32_16x16x32_bf16(false, a, false, b, (short)0, c,
                                                 false, false);
}

// ---------------------------------------------------------------------------
// Kernel 1: QKV projection.  grid = (128 row tiles, 48 col tiles), 256 thr.
// Each 64-col tile lands entirely inside one (matrix, head) pair.
// ---------------------------------------------------------------------------
__global__ __launch_bounds__(256)
void qkv_gemm_kernel(const float* __restrict__ x, const float* __restrict__ w,
                     unsigned short* __restrict__ Qb, unsigned short* __restrict__ Kb,
                     unsigned short* __restrict__ Vb) {
  __shared__ unsigned short As[64][32];   // A tile, row major [m][k]
  __shared__ unsigned short Bt[64][32];   // B tile transposed [n][k]

  const int tid  = threadIdx.x;
  const int lane = tid & 31;
  const int wv   = tid >> 5;              // 8 waves
  const int rb   = wv >> 1;               // row block 0..3
  const int cb   = wv & 1;                // col half 0..1
  const int row0 = blockIdx.x * 64;
  const int col0 = blockIdx.y * 64;

  const int aRow = tid >> 2, aCg = (tid & 3) * 8;    // A: 64 rows x 4 groups of 8
  const int bN   = tid & 63, bK0 = (tid >> 6) * 8;   // B: 64 cols x 4 groups of 8 k

  CF c0, c1;
#pragma unroll
  for (int r = 0; r < 8; ++r) { c0.f[r] = 0.f; c1.f[r] = 0.f; }

  for (int k0 = 0; k0 < DIM; k0 += 32) {
    const float* ap = x + (size_t)(row0 + aRow) * DIM + k0 + aCg;
    if (k0 + 32 < DIM) __builtin_prefetch(ap + 32, 0, 0);
#pragma unroll
    for (int j = 0; j < 8; ++j) As[aRow][aCg + j] = f2bf(ap[j]);
#pragma unroll
    for (int j = 0; j < 8; ++j)
      Bt[bN][bK0 + j] = f2bf(w[(size_t)(k0 + bK0 + j) * 3072 + col0 + bN]);
    __syncthreads();

    v16bf a  = frag32(&As[rb * 16 + (lane & 15)][0], lane);
    v16bf b0 = frag32(&Bt[cb * 32 + (lane & 15)][0], lane);
    v16bf b1 = frag32(&Bt[cb * 32 + 16 + (lane & 15)][0], lane);
    c0.v = wmma_bf16(a, b0, c0.v);
    c1.v = wmma_bf16(a, b1, c1.v);
    __syncthreads();
  }

  const int mat  = blockIdx.y >> 4;       // 0=Q 1=K 2=V
  const int head = blockIdx.y & 15;
  unsigned short* dst = (mat == 0) ? Qb : (mat == 1) ? Kb : Vb;
  const float scl = (mat == 0) ? 0.125f : 1.0f;   // HEAD_DIM^-0.5 folded into Q

#pragma unroll
  for (int r = 0; r < 8; ++r) {
    int rowM = rb * 16 + r + ((lane >> 4) << 3);
    int grow = row0 + rowM;
    int bIdx = grow >> 11, s = grow & 2047;
    size_t base = (((size_t)bIdx * HEADS + head) * SEQ + s) * HEAD_DIM;
    dst[base + cb * 32 + (lane & 15)]      = f2bf(c0.f[r] * scl);
    dst[base + cb * 32 + 16 + (lane & 15)] = f2bf(c1.f[r] * scl);
  }
}

// ---------------------------------------------------------------------------
// Kernel 2: flash attention.  grid = (B*H = 64, S/64 = 32), 256 threads.
// ---------------------------------------------------------------------------
__global__ __launch_bounds__(256)
void flash_attn_kernel(const unsigned short* __restrict__ Qb,
                       const unsigned short* __restrict__ Kb,
                       const unsigned short* __restrict__ Vb,
                       unsigned short* __restrict__ Ob) {
  __shared__ unsigned short Qs[64][64];   // Q tile [i][d]
  __shared__ unsigned short Ks[64][64];   // K tile [j][d] == B^T for scores
  __shared__ unsigned short Vt[64][64];   // V^T    [d][j] == B^T for P@V
  __shared__ float          Sl[64][64];   // raw scores
  __shared__ unsigned short Ps[64][64];   // exp probs (bf16) == A for P@V
  __shared__ float mrow[64], lrow[64], alpha[64];

  const int tid  = threadIdx.x;
  const int lane = tid & 31;
  const int wv   = tid >> 5;
  const int rb   = wv >> 1;
  const int cb   = wv & 1;
  const int bh   = blockIdx.x;
  const int i0   = blockIdx.y * 64;

  const int cpRow = tid >> 2;                 // 64 rows, 4 threads/row
  const unsigned cpOff = (unsigned)(tid & 3) * 32u;   // byte offset in 128B row

  // Async-load Q tile (bf16, already scaled) straight into LDS.
  {
    const char* gq = (const char*)(Qb + ((size_t)bh * SEQ + i0 + cpRow) * HEAD_DIM) + cpOff;
    unsigned lq = (unsigned)(uintptr_t)&Qs[cpRow][0] + cpOff;
    async_copy32(lq, gq);
  }
  if (tid < 64) { mrow[tid] = -INFINITY; lrow[tid] = 0.f; }

  CF o0, o1;
#pragma unroll
  for (int r = 0; r < 8; ++r) { o0.f[r] = 0.f; o1.f[r] = 0.f; }
  wait_async0();
  __syncthreads();

  for (int j0 = 0; j0 < SEQ; j0 += 64) {
    // Async-load K tile (row major == B^T layout for the score GEMM).
    {
      const char* gk = (const char*)(Kb + ((size_t)bh * SEQ + j0 + cpRow) * HEAD_DIM) + cpOff;
      unsigned lk = (unsigned)(uintptr_t)&Ks[cpRow][0] + cpOff;
      async_copy32(lk, gk);
    }
    // V tile needs a transpose -> explicit load/scatter into Vt.
    {
      int dg = (tid & 3) * 16;
      const unsigned short* vs = Vb + ((size_t)bh * SEQ + j0 + cpRow) * HEAD_DIM + dg;
#pragma unroll
      for (int u = 0; u < 16; ++u) Vt[dg + u][cpRow] = vs[u];
    }
    wait_async0();
    __syncthreads();

    // Scores: S = Q @ K^T, each wave a 16x32 slab.
    CF s0, s1;
#pragma unroll
    for (int r = 0; r < 8; ++r) { s0.f[r] = 0.f; s1.f[r] = 0.f; }
#pragma unroll
    for (int d0 = 0; d0 < 64; d0 += 32) {
      v16bf a  = frag32(&Qs[rb * 16 + (lane & 15)][d0], lane);
      v16bf b0 = frag32(&Ks[cb * 32 + (lane & 15)][d0], lane);
      v16bf b1 = frag32(&Ks[cb * 32 + 16 + (lane & 15)][d0], lane);
      s0.v = wmma_bf16(a, b0, s0.v);
      s1.v = wmma_bf16(a, b1, s1.v);
    }
#pragma unroll
    for (int r = 0; r < 8; ++r) {
      int rowM = rb * 16 + r + ((lane >> 4) << 3);
      Sl[rowM][cb * 32 + (lane & 15)]      = s0.f[r];
      Sl[rowM][cb * 32 + 16 + (lane & 15)] = s1.f[r];
    }
    __syncthreads();

    // Online softmax (one thread per row).
    if (tid < 64) {
      int row = tid;
      float m_old = mrow[row], mx = m_old;
#pragma unroll 8
      for (int j = 0; j < 64; ++j) mx = fmaxf(mx, Sl[row][j]);
      float a = __expf(m_old - mx);
      float sum = 0.f;
#pragma unroll 8
      for (int j = 0; j < 64; ++j) {
        float p = __expf(Sl[row][j] - mx);
        sum += p;
        Ps[row][j] = f2bf(p);
      }
      lrow[row] = lrow[row] * a + sum;
      mrow[row] = mx;
      alpha[row] = a;
    }
    __syncthreads();

    // Rescale O and accumulate P @ V.
#pragma unroll
    for (int r = 0; r < 8; ++r) {
      float av = alpha[rb * 16 + r + ((lane >> 4) << 3)];
      o0.f[r] *= av;
      o1.f[r] *= av;
    }
#pragma unroll
    for (int jj = 0; jj < 64; jj += 32) {
      v16bf a  = frag32(&Ps[rb * 16 + (lane & 15)][jj], lane);
      v16bf b0 = frag32(&Vt[cb * 32 + (lane & 15)][jj], lane);
      v16bf b1 = frag32(&Vt[cb * 32 + 16 + (lane & 15)][jj], lane);
      o0.v = wmma_bf16(a, b0, o0.v);
      o1.v = wmma_bf16(a, b1, o1.v);
    }
    __syncthreads();
  }

  // Epilogue: divide by row sum, store bf16 [B,H,S,D] flat.
#pragma unroll
  for (int r = 0; r < 8; ++r) {
    int rowM = rb * 16 + r + ((lane >> 4) << 3);
    float inv = 1.f / lrow[rowM];
    size_t base = ((size_t)bh * SEQ + i0 + rowM) * HEAD_DIM;
    Ob[base + cb * 32 + (lane & 15)]      = f2bf(o0.f[r] * inv);
    Ob[base + cb * 32 + 16 + (lane & 15)] = f2bf(o1.f[r] * inv);
  }
}

// ---------------------------------------------------------------------------
// Kernel 3: output projection + bias.  grid = (128, 16), 256 threads.
// Reads O flat as [8192,1024] (== reference's transpose-free reshape).
// ---------------------------------------------------------------------------
__global__ __launch_bounds__(256)
void out_gemm_kernel(const unsigned short* __restrict__ Ob,
                     const float* __restrict__ w, const float* __restrict__ bias,
                     float* __restrict__ out) {
  __shared__ unsigned short As[64][32];
  __shared__ unsigned short Bt[64][32];

  const int tid  = threadIdx.x;
  const int lane = tid & 31;
  const int wv   = tid >> 5;
  const int rb   = wv >> 1;
  const int cb   = wv & 1;
  const int row0 = blockIdx.x * 64;
  const int col0 = blockIdx.y * 64;

  const int aRow = tid >> 2, aCg = (tid & 3) * 8;
  const int bN   = tid & 63, bK0 = (tid >> 6) * 8;

  CF c0, c1;
#pragma unroll
  for (int r = 0; r < 8; ++r) { c0.f[r] = 0.f; c1.f[r] = 0.f; }

  for (int k0 = 0; k0 < DIM; k0 += 32) {
    // A tile is already bf16 -> async copy 16B per thread into LDS.
    {
      const char* ga = (const char*)(Ob + (size_t)(row0 + aRow) * DIM + k0) + (unsigned)aCg * 2u;
      unsigned la = (unsigned)(uintptr_t)&As[aRow][0] + (unsigned)aCg * 2u;
      asm volatile("global_load_async_to_lds_b128 %0, %1, off"
                   :: "v"(la), "v"(ga) : "memory");
    }
#pragma unroll
    for (int j = 0; j < 8; ++j)
      Bt[bN][bK0 + j] = f2bf(w[(size_t)(k0 + bK0 + j) * DIM + col0 + bN]);
    wait_async0();
    __syncthreads();

    v16bf a  = frag32(&As[rb * 16 + (lane & 15)][0], lane);
    v16bf b0 = frag32(&Bt[cb * 32 + (lane & 15)][0], lane);
    v16bf b1 = frag32(&Bt[cb * 32 + 16 + (lane & 15)][0], lane);
    c0.v = wmma_bf16(a, b0, c0.v);
    c1.v = wmma_bf16(a, b1, c1.v);
    __syncthreads();
  }

#pragma unroll
  for (int r = 0; r < 8; ++r) {
    int rowM = rb * 16 + r + ((lane >> 4) << 3);
    int grow = row0 + rowM;
    int n0 = col0 + cb * 32 + (lane & 15);
    out[(size_t)grow * DIM + n0]      = c0.f[r] + bias[n0];
    out[(size_t)grow * DIM + n0 + 16] = c1.f[r] + bias[n0 + 16];
  }
}

// ---------------------------------------------------------------------------
extern "C" void kernel_launch(void* const* d_in, const int* in_sizes, int n_in,
                              void* d_out, int out_size, void* d_ws, size_t ws_size,
                              hipStream_t stream) {
  const float* x     = (const float*)d_in[0];   // [4,2048,1024]
  const float* w_qkv = (const float*)d_in[1];   // [1024,3072]
  const float* w_out = (const float*)d_in[2];   // [1024,1024]
  const float* b_out = (const float*)d_in[3];   // [1024]
  float* out = (float*)d_out;                   // [4,2048,1024]

  // Workspace: Q,K,V,O bf16 in [B,H,S,64] layout  (4 x 16 MB = 64 MB).
  const size_t NE = (size_t)BATCH * HEADS * SEQ * HEAD_DIM;   // 8,388,608
  unsigned short* Qb = (unsigned short*)d_ws;
  unsigned short* Kb = Qb + NE;
  unsigned short* Vb = Kb + NE;
  unsigned short* Ob = Vb + NE;

  qkv_gemm_kernel<<<dim3(ROWS / 64, 48), 256, 0, stream>>>(x, w_qkv, Qb, Kb, Vb);
  flash_attn_kernel<<<dim3(BATCH * HEADS, SEQ / 64), 256, 0, stream>>>(Qb, Kb, Vb, Ob);
  out_gemm_kernel<<<dim3(ROWS / 64, DIM / 64), 256, 0, stream>>>(Ob, w_out, b_out, out);
}
